// VectorQuantizer_52390011076664
// MI455X (gfx1250) — matrix operand order repaired and checked by reading, not compiled, not run on hardware
//
#include <hip/hip_runtime.h>
#include <hip/hip_bf16.h>
#include <stdint.h>

#define B_    16
#define C_    192
#define L_    4096
#define K_    1024
#define WAVES 8
#define LTILE 16
#define KSTAGE 32
#define NSTAGE (K_ / KSTAGE)
#define LSTRIDE 200   // padded LDS row stride (bf16 elems): 400B = 36 dwords mod 64 banks

typedef __bf16 v16bf __attribute__((ext_vector_type(16)));
typedef __bf16 v8bf  __attribute__((ext_vector_type(8)));
typedef float  v8f   __attribute__((ext_vector_type(8)));
typedef float  v4f   __attribute__((ext_vector_type(4)));
typedef unsigned int v2u __attribute__((ext_vector_type(2)));

static __device__ __forceinline__ unsigned short f2bf(float f) {
  unsigned u = __float_as_uint(f);
  unsigned r = ((u >> 16) & 1u) + 0x7FFFu;      // round-to-nearest-even
  return (unsigned short)((u + r) >> 16);
}

// branchless sorted-top4 insert of packed (value|index) float
#define TOP4_INSERT(cand)                                     \
  do {                                                        \
    float _c = (cand), _m;                                    \
    _m = fmaxf(t0, _c); _c = fminf(t0, _c); t0 = _m;          \
    _m = fmaxf(t1, _c); _c = fminf(t1, _c); t1 = _m;          \
    _m = fmaxf(t2, _c); _c = fminf(t2, _c); t2 = _m;          \
    t3 = fmaxf(t3, _c);                                       \
  } while (0)

__global__ __launch_bounds__(256)
void vq_topk_wmma_kernel(const float* __restrict__ x,
                         const int*   __restrict__ sid,
                         const float* __restrict__ codebooks,
                         float* __restrict__ out) {
  __shared__ __align__(16) unsigned short s_codes[KSTAGE * LSTRIDE];        // 12.5 KB
  __shared__ __align__(16) unsigned short s_q[WAVES * LTILE * LSTRIDE];     // 50.0 KB

  const int tid  = threadIdx.x;
  const int wave = tid >> 5;
  const int lane = tid & 31;
  const int j    = lane & 15;   // l column within wave tile / k row within A tile
  const int half = lane >> 4;   // lane half selects K-sub-range per ISA layouts

  const int b    = blockIdx.x >> 5;                       // 16 batches
  const int lblk = (blockIdx.x & 31) * (WAVES * LTILE);   // 128 l per workgroup
  const int l    = lblk + wave * LTILE + j;

  const float* xb = x + (size_t)b * C_ * L_;
  const float* cb = codebooks + (size_t)sid[b] * K_ * C_;

  // ---------- phase 1: L2-normalize q columns, write bf16 tile to LDS ----------
  const int cbeg = half * (C_ / 2);
  float ss = 0.f;
  #pragma unroll 4
  for (int c = 0; c < C_ / 2; ++c) {
    float v = xb[(size_t)(cbeg + c) * L_ + l];
    ss += v * v;
  }
  ss += __shfl_xor(ss, 16, 32);
  const float inv = 1.0f / fmaxf(sqrtf(ss), 1e-6f);
  unsigned short* qrow = &s_q[(wave * LTILE + j) * LSTRIDE];
  #pragma unroll 4
  for (int c = 0; c < C_ / 2; c += 2) {
    float a0 = xb[(size_t)(cbeg + c)     * L_ + l] * inv;
    float a1 = xb[(size_t)(cbeg + c + 1) * L_ + l] * inv;
    *(unsigned*)&qrow[cbeg + c] =
        (unsigned)f2bf(a0) | ((unsigned)f2bf(a1) << 16);
  }

  // ---------- staging prefetch setup (hoisted address math) ----------
  const float*  gptr[6];
  unsigned      sofs[6];
  #pragma unroll
  for (int it = 0; it < 6; ++it) {
    int g   = it * 256 + tid;
    int row = g / (C_ / 4);
    int c4  = (g % (C_ / 4)) * 4;
    gptr[it] = cb + (size_t)row * C_ + c4;
    sofs[it] = row * LSTRIDE + c4;
  }
  v4f raw[6];
  #pragma unroll
  for (int it = 0; it < 6; ++it) {            // prefetch stage 0
    raw[it] = *(const v4f*)gptr[it];
    gptr[it] += KSTAGE * C_;
  }

  __syncthreads();   // q tile visible

  // ---------- B operands (q, 32xK16 per chunk): hoisted, reused for all k ----------
  // B layout (32x16 bf16): lane n<16 holds col N=n, K={0..7,16..23}; lane n+16 K={8..15,24..31}
  v16bf bq[6];
  #pragma unroll
  for (int cc = 0; cc < 6; ++cc) {
    const unsigned short* p =
        &s_q[(wave * LTILE + j) * LSTRIDE + cc * 32 + half * 8];
    union { v16bf v; v8bf h[2]; } u;
    u.h[0] = *(const v8bf*)(p);
    u.h[1] = *(const v8bf*)(p + 16);
    bq[cc] = u.v;
  }

  const float NEG = __int_as_float(0xFF800000);  // -inf
  float t0 = NEG, t1 = NEG, t2 = NEG, t3 = NEG;

  // ---------- main loop: pipelined stage of 32 codebook rows, 2 WMMA k-tiles ----------
  for (int stage = 0; stage < NSTAGE; ++stage) {
    __syncthreads();   // previous-stage readers done
    #pragma unroll
    for (int it = 0; it < 6; ++it) {          // cvt+store prefetched rows
      v2u pk;
      pk.x = (unsigned)f2bf(raw[it][0]) | ((unsigned)f2bf(raw[it][1]) << 16);
      pk.y = (unsigned)f2bf(raw[it][2]) | ((unsigned)f2bf(raw[it][3]) << 16);
      *(v2u*)&s_codes[sofs[it]] = pk;
    }
    __syncthreads();

    if (stage + 1 < NSTAGE) {                 // issue next stage's loads; wait deferred
      #pragma unroll
      for (int it = 0; it < 6; ++it) {
        raw[it] = *(const v4f*)gptr[it];
        gptr[it] += KSTAGE * C_;
      }
    }

    #pragma unroll
    for (int kt = 0; kt < KSTAGE / 16; ++kt) {
      v8f acc = {};
      #pragma unroll
      for (int cc = 0; cc < 6; ++cc) {
        // A layout (16x32 bf16): lane m<16 row M=m, K={0..7,16..23}; lane m+16 K={8..15,24..31}
        const unsigned short* p =
            &s_codes[(kt * 16 + j) * LSTRIDE + cc * 32 + half * 8];
        union { v16bf v; v8bf h[2]; } u;
        u.h[0] = *(const v8bf*)(p);
        u.h[1] = *(const v8bf*)(p + 16);
        acc = __builtin_amdgcn_wmma_f32_16x16x32_bf16(
            false, u.v, false, bq[cc], (short)0, acc, false, false);
      }
      // C/D layout: VGPR r, lane<16 -> M=r ; lane>=16 -> M=8+r ; N = lane%16 = l col
      const int kbase = stage * KSTAGE + kt * 16 + half * 8;
      #pragma unroll
      for (int r = 0; r < 8; ++r) {
        unsigned bits = __float_as_uint(acc[r]);
        float cand = __uint_as_float((bits & 0xFFFFFC00u) | (unsigned)(kbase + r));
        TOP4_INSERT(cand);
      }
    }
  }

  // ---------- merge lane-pair top-4 (each lane saw half the k space) ----------
  float o0 = __shfl_xor(t0, 16, 32);
  float o1 = __shfl_xor(t1, 16, 32);
  float o2 = __shfl_xor(t2, 16, 32);
  float o3 = __shfl_xor(t3, 16, 32);
  TOP4_INSERT(o0); TOP4_INSERT(o1); TOP4_INSERT(o2); TOP4_INSERT(o3);

  const int k0 = __float_as_uint(t0) & 1023;
  const int k1 = __float_as_uint(t1) & 1023;
  const int k2 = __float_as_uint(t2) & 1023;
  const int k3 = __float_as_uint(t3) & 1023;

  // ---------- gather selected f32 rows, average, NT-store out[b,c,l] ----------
  const float* r0 = cb + (size_t)k0 * C_;
  const float* r1 = cb + (size_t)k1 * C_;
  const float* r2 = cb + (size_t)k2 * C_;
  const float* r3 = cb + (size_t)k3 * C_;
  float* ob = out + (size_t)b * C_ * L_ + l;
  #pragma unroll 4
  for (int c = cbeg; c < cbeg + C_ / 2; c += 4) {
    v4f a = (*(const v4f*)(r0 + c) + *(const v4f*)(r1 + c) +
             *(const v4f*)(r2 + c) + *(const v4f*)(r3 + c)) * 0.25f;
    #pragma unroll
    for (int i = 0; i < 4; ++i)
      __builtin_nontemporal_store(a[i], &ob[(size_t)(c + i) * L_]);
  }
}

extern "C" void kernel_launch(void* const* d_in, const int* in_sizes, int n_in,
                              void* d_out, int out_size, void* d_ws, size_t ws_size,
                              hipStream_t stream) {
  const float* x         = (const float*)d_in[0];
  const int*   sid       = (const int*)d_in[1];
  const float* codebooks = (const float*)d_in[2];
  float*       out       = (float*)d_out;
  (void)in_sizes; (void)n_in; (void)out_size; (void)d_ws; (void)ws_size;

  dim3 grid(B_ * (L_ / (WAVES * LTILE)));   // 16 * 32 = 512 workgroups
  dim3 block(WAVES * 32);                   // 8 waves of 32
  vq_topk_wmma_kernel<<<grid, block, 0, stream>>>(x, sid, codebooks, out);
}
